// MultiHeadAttention_75007308858133
// MI455X (gfx1250) — matrix operand order repaired and checked by reading, not compiled, not run on hardware
//
#include <hip/hip_runtime.h>

// ---------------------------------------------------------------------------
// MHA forward for MI455X (gfx1250, wave32, WMMA bf16 16x16x32, f32 accum).
//  - weights transposed [N x K], x/concat packed in A-tile order, V as [dv x S]
//    => every fragment load is 16 contiguous bf16 per lane (b128 pairs)
//  - flash attention: K/V tiles double-buffered in LDS with bank-conflict-free
//    padded strides; all fragments preloaded into distinct registers so
//    ds_load waits stagger instead of serializing each WMMA
//  - softmax row-sums via P @ ones WMMA; GEMM: 64x32 tile per wave
// ---------------------------------------------------------------------------

#define S_LEN   4096
#define DMODEL  512
#define NHEAD   8
#define DKH     128
#define DVH     64
#define KSTRIDE 136   // 32x128 K tile rows padded: 272B = 68 words -> banks 4n
#define VSTRIDE 40    // 64x32  V tile rows padded: 80B  = 20 words -> all banks

typedef __bf16 bf16;
typedef bf16  v16bf __attribute__((ext_vector_type(16)));
typedef bf16  v8bf  __attribute__((ext_vector_type(8)));
typedef float v8f   __attribute__((ext_vector_type(8)));

static __device__ __forceinline__ v8f wmma_bf16(v16bf a, v16bf b, v8f c) {
  return __builtin_amdgcn_wmma_f32_16x16x32_bf16(false, a, false, b, (short)0, c,
                                                 false, false);
}

// 16 contiguous bf16 (32B, >=16B-aligned at all call sites) -> 2 x b128.
static __device__ __forceinline__ v16bf load16(const bf16* p) {
  const v8bf* q = (const v8bf*)p;
  v8bf lo = q[0];
  v8bf hi = q[1];
  return __builtin_shufflevector(lo, hi, 0, 1, 2, 3, 4, 5, 6, 7,
                                 8, 9, 10, 11, 12, 13, 14, 15);
}

static __device__ __forceinline__ void store16(bf16* p, v16bf f) {
  v8bf lo = __builtin_shufflevector(f, f, 0, 1, 2, 3, 4, 5, 6, 7);
  v8bf hi = __builtin_shufflevector(f, f, 8, 9, 10, 11, 12, 13, 14, 15);
  ((v8bf*)p)[0] = lo;
  ((v8bf*)p)[1] = hi;
}

// Row-major A-frag loader (used once per wave for Q; 8 x b32 per lane).
static __device__ __forceinline__ v16bf load_a_frag_rm(const bf16* __restrict__ A,
                                                       int lda) {
  const int lane = threadIdx.x & 31;
  const int m    = lane & 15;
  const int half = lane >> 4;
  const bf16* row = A + (long)m * lda;
  v16bf f;
#pragma unroll
  for (int v = 0; v < 8; ++v) {
    const int kb = half * 16 + (v << 1);
    f[2 * v]     = row[kb];
    f[2 * v + 1] = row[kb + 1];
  }
  return f;
}

// ---------------------------------------------------------------------------
// Layout-transform cast kernels.

// out[h][n][k] = (bf16) in[h][k][n]   (weights -> transposed bf16)
__global__ void cast_transpose(const float* __restrict__ in, bf16* __restrict__ out,
                               int K, int N, long total) {
  const long idx = (long)blockIdx.x * blockDim.x + threadIdx.x;
  if (idx >= total) return;
  const int  k = (int)(idx % K);
  const int  n = (int)((idx / K) % N);
  const long h = idx / ((long)K * N);
  out[idx] = (bf16)in[(h * K + k) * (long)N + n];
}

// Pack a row-major [M x K] f32 matrix into A-fragment tile order:
// tile t = m16tile*(K/32)+ktile, 512 bf16 per tile laid out [lane][16].
__global__ void cast_pack_a(const float* __restrict__ in, bf16* __restrict__ out,
                            int K, long total) {
  const long idx = (long)blockIdx.x * blockDim.x + threadIdx.x;
  if (idx >= total) return;
  const int  e    = (int)(idx & 15);
  const int  lane = (int)((idx >> 4) & 31);
  const long tile = idx >> 9;
  const int  kt   = K >> 5;
  const long mt   = tile / kt;
  const int  ktl  = (int)(tile % kt);
  const int  v = e >> 1, ee = e & 1;
  const int  m = lane & 15, half = lane >> 4;
  const int  kk = half * 16 + (v << 1) + ee;
  out[idx] = (bf16)in[(mt * 16 + m) * (long)K + ktl * 32 + kk];
}

// ---------------------------------------------------------------------------
// Batched GEMM: C[h] = A @ B[h] + bias[h].
// A: packed A-tile bf16 [M x K].  BT: transposed bf16 [N x K] per head.
// 64x32 output tile per wave.  TRANS_C=true stores C as [N x M] (for V^T).
template <bool TRANS_C, typename OutT>
__global__ void __launch_bounds__(256) gemm_wmma(
    const bf16* __restrict__ A, const bf16* __restrict__ BT,
    const float* __restrict__ bias, OutT* __restrict__ C,
    int M, int N, int K, long sB, long sC, long sBias) {
  const int h = blockIdx.y;
  BT += (long)h * sB;
  C  += (long)h * sC;
  const float* biash = bias + (long)h * sBias;

  const int wave   = blockIdx.x * (blockDim.x >> 5) + (threadIdx.x >> 5);
  const int tilesN = N >> 5;
  const int tilesM = M >> 6;
  if (wave >= tilesM * tilesN) return;  // wave-uniform; EXEC stays all-1s
  const int tm = wave / tilesN;
  const int tn = wave % tilesN;

  const int lane = threadIdx.x & 31;
  const int nn   = lane & 15;
  const int hf   = lane >> 4;
  const int kt   = K >> 5;

  v8f acc[4][2] = {};
  const bf16* Ap[4];
#pragma unroll
  for (int i = 0; i < 4; ++i)
    Ap[i] = A + (((long)(4 * tm + i) * kt) << 9) + lane * 16;
  const bf16* Bp0 = BT + (long)(tn * 32 + nn) * K + hf * 16;
  const bf16* Bp1 = Bp0 + (long)16 * K;

  for (int k = 0; k < K; k += 32) {
    v16bf b0 = load16(Bp0 + k);
    v16bf b1 = load16(Bp1 + k);
    v16bf a0 = load16(Ap[0] + (k << 4));  // (k/32)*512
    v16bf a1 = load16(Ap[1] + (k << 4));
    v16bf a2 = load16(Ap[2] + (k << 4));
    v16bf a3 = load16(Ap[3] + (k << 4));
    acc[0][0] = wmma_bf16(a0, b0, acc[0][0]);
    acc[0][1] = wmma_bf16(a0, b1, acc[0][1]);
    acc[1][0] = wmma_bf16(a1, b0, acc[1][0]);
    acc[1][1] = wmma_bf16(a1, b1, acc[1][1]);
    acc[2][0] = wmma_bf16(a2, b0, acc[2][0]);
    acc[2][1] = wmma_bf16(a2, b1, acc[2][1]);
    acc[3][0] = wmma_bf16(a3, b0, acc[3][0]);
    acc[3][1] = wmma_bf16(a3, b1, acc[3][1]);
  }

  const float bv0 = biash[tn * 32 + nn];
  const float bv1 = biash[tn * 32 + 16 + nn];
  const int col0 = tn * 32 + nn;
  const int col1 = col0 + 16;
#pragma unroll
  for (int i = 0; i < 4; ++i) {
#pragma unroll
    for (int r = 0; r < 8; ++r) {
      const int row = tm * 64 + i * 16 + hf * 8 + r;  // C layout: M = r + 8*half
      if (TRANS_C) {
        C[(long)col0 * M + row] = (OutT)(acc[i][0][r] + bv0);
        C[(long)col1 * M + row] = (OutT)(acc[i][1][r] + bv1);
      } else {
        C[(long)row * N + col0] = (OutT)(acc[i][0][r] + bv0);
        C[(long)row * N + col1] = (OutT)(acc[i][1][r] + bv1);
      }
    }
  }
}

// ---------------------------------------------------------------------------
// Flash attention: grid (S/128, H), block 256 (8 waves). Each wave owns 16
// query rows. K/V tiles for each 32-key block staged once per workgroup into
// double-buffered, bank-conflict-free LDS. Row-sums via P @ ones WMMA.
__global__ void __launch_bounds__(256) flash_attn_wmma(
    const bf16* __restrict__ Q, const bf16* __restrict__ Kmat,
    const bf16* __restrict__ Vt, bf16* __restrict__ Op) {
  const int h = blockIdx.y;
  const bf16* Qh  = Q    + (long)h * S_LEN * DKH;
  const bf16* Kh  = Kmat + (long)h * S_LEN * DKH;
  const bf16* Vth = Vt   + (long)h * DVH * S_LEN;  // [DVH x S]

  const int tid  = threadIdx.x;
  const int wave = tid >> 5;
  const int lane = tid & 31;
  const int nn   = lane & 15;
  const int hf   = lane >> 4;
  const int q0   = blockIdx.x * 128 + wave * 16;

  __shared__ bf16 kbuf[2][32 * KSTRIDE];  // padded rows -> conflict-free b128
  __shared__ bf16 vbuf[2][DVH * VSTRIDE];
  __shared__ bf16 pbuf[8][16 * 32];       // per-wave bounce
  bf16* myp = pbuf[wave];

  // C-frag (m,kk) -> packed-A destination helpers (kk = nn / 16+nn):
  const int bb = 0;
  const int e0 = nn;

  v16bf qf[4];
#pragma unroll
  for (int c = 0; c < 4; ++c)
    qf[c] = load_a_frag_rm(Qh + (long)q0 * DKH + c * 32, DKH);

  v16bf ones;
#pragma unroll
  for (int i = 0; i < 16; ++i) ones[i] = (bf16)1.0f;

  v8f o0 = {}, o1 = {}, o2 = {}, o3 = {};
  v8f lacc = {};                       // online row-sum accumulator
  float mrow[8];
#pragma unroll
  for (int r = 0; r < 8; ++r) mrow[r] = -1e30f;

  const float scale = 0.044194173824159216f;  // 1/sqrt(D_MODEL=512)

  // Stage one 32-key block (K: 32x128, V: 64x32) into LDS buffer `buf`.
  auto stage = [&](int jb, int buf) {
    const int j = jb * 32;
    {  // K tile: each thread copies 32 contiguous bytes
      const int row = tid >> 3, col = (tid & 7) * 16;
      const v8bf* g = (const v8bf*)(Kh + (long)(j + row) * DKH + col);
      v8bf x0 = g[0], x1 = g[1];
      v8bf* l = (v8bf*)(&kbuf[buf][row * KSTRIDE + col]);
      l[0] = x0;
      l[1] = x1;
    }
    {  // V tile: each thread copies 16 contiguous bytes
      const int d = tid >> 2, c4 = (tid & 3) * 8;
      *(v8bf*)(&vbuf[buf][d * VSTRIDE + c4]) =
          *(const v8bf*)(Vth + (long)d * S_LEN + j + c4);
    }
  };

  stage(0, 0);
  __syncthreads();

  const int nblocks = S_LEN / 32;
  for (int jb = 0; jb < nblocks; ++jb) {
    const int buf = jb & 1;
    if (jb + 1 < nblocks) stage(jb + 1, buf ^ 1);
    if (jb + 2 < nblocks)
      __builtin_prefetch(Kh + (long)(jb + 2) * 32 * DKH, 0, 1);  // global_prefetch_b8

    const bf16* kb = kbuf[buf];
    const bf16* vb = vbuf[buf];

    // Preload all 8 K fragments (distinct registers -> staggered dscnt waits).
    v16bf kf[8];
#pragma unroll
    for (int c = 0; c < 4; ++c) {
      kf[2 * c]     = load16(kb + (nn)      * KSTRIDE + c * 32 + hf * 16);
      kf[2 * c + 1] = load16(kb + (16 + nn) * KSTRIDE + c * 32 + hf * 16);
    }
    // S = Q(16x128) @ K^T(128x32): 8 back-to-back WMMAs on two accumulators.
    v8f s0 = {}, s1 = {};
#pragma unroll
    for (int c = 0; c < 4; ++c) {
      s0 = wmma_bf16(qf[c], kf[2 * c],     s0);
      s1 = wmma_bf16(qf[c], kf[2 * c + 1], s1);
    }
    // Online max (cross-lane xor reduction within 16-lane halves).
    float alpha[8];
#pragma unroll
    for (int r = 0; r < 8; ++r) {
      const float x0 = s0[r] * scale;
      const float x1 = s1[r] * scale;
      float mx = fmaxf(x0, x1);
      mx = fmaxf(mx, __shfl_xor(mx, 1, 32));
      mx = fmaxf(mx, __shfl_xor(mx, 2, 32));
      mx = fmaxf(mx, __shfl_xor(mx, 4, 32));
      mx = fmaxf(mx, __shfl_xor(mx, 8, 32));
      const float mnew = fmaxf(mrow[r], mx);
      alpha[r] = __expf(mrow[r] - mnew);
      mrow[r]  = mnew;
      s0[r] = __expf(x0 - mnew);
      s1[r] = __expf(x1 - mnew);
    }
    // Issue V fragment loads early (K frags are dead; overlap with scaling).
    v16bf vf0 = load16(vb + (nn)      * VSTRIDE + hf * 16);
    v16bf vf1 = load16(vb + (16 + nn) * VSTRIDE + hf * 16);
    v16bf vf2 = load16(vb + (32 + nn) * VSTRIDE + hf * 16);
    v16bf vf3 = load16(vb + (48 + nn) * VSTRIDE + hf * 16);
#pragma unroll
    for (int r = 0; r < 8; ++r) {
      o0[r] *= alpha[r]; o1[r] *= alpha[r];
      o2[r] *= alpha[r]; o3[r] *= alpha[r];
      lacc[r] *= alpha[r];
    }
    // P (16x32) -> packed A order in LDS, then one contiguous 32B read/lane.
#pragma unroll
    for (int r = 0; r < 8; ++r) {
      const int lp = (hf * 8 + r + bb) * 16 + e0;
      myp[lp]     = (bf16)s0[r];
      myp[lp + 256] = (bf16)s1[r];
    }
    v16bf pf = load16(myp + lane * 16);
    // Row sums: lacc += P @ ones  (value replicated across the 16 columns).
    lacc = wmma_bf16(pf, ones, lacc);
    // O += P @ V(32x64).
    o0 = wmma_bf16(pf, vf0, o0);
    o1 = wmma_bf16(pf, vf1, o1);
    o2 = wmma_bf16(pf, vf2, o2);
    o3 = wmma_bf16(pf, vf3, o3);

    __syncthreads();  // all reads of buf done; staging of buf^1 visible
  }

  // Normalize, emit two 16x32 output tiles (cols h*64..h*64+63) in packed
  // A-tile order via the same per-wave LDS bounce.
  float inv[8];
#pragma unroll
  for (int r = 0; r < 8; ++r) inv[r] = 1.0f / lacc[r];

  const long tileBase = (long)(q0 >> 4) * (DMODEL / 32);
#pragma unroll
  for (int r = 0; r < 8; ++r) {
    const int lp = (hf * 8 + r + bb) * 16 + e0;
    myp[lp]     = (bf16)(o0[r] * inv[r]);
    myp[lp + 256] = (bf16)(o1[r] * inv[r]);
  }
  store16(Op + ((tileBase + 2 * h) << 9) + lane * 16, load16(myp + lane * 16));
#pragma unroll
  for (int r = 0; r < 8; ++r) {
    const int lp = (hf * 8 + r + bb) * 16 + e0;
    myp[lp]     = (bf16)(o2[r] * inv[r]);
    myp[lp + 256] = (bf16)(o3[r] * inv[r]);
  }
  store16(Op + ((tileBase + 2 * h + 1) << 9) + lane * 16, load16(myp + lane * 16));
}

// ---------------------------------------------------------------------------
extern "C" void kernel_launch(void* const* d_in, const int* in_sizes, int n_in,
                              void* d_out, int out_size, void* d_ws, size_t ws_size,
                              hipStream_t stream) {
  (void)in_sizes; (void)n_in; (void)out_size; (void)ws_size;
  const float* x  = (const float*)d_in[0];
  const float* Wq = (const float*)d_in[1];
  const float* bq = (const float*)d_in[2];
  const float* Wk = (const float*)d_in[3];
  const float* bk = (const float*)d_in[4];
  const float* Wv = (const float*)d_in[5];
  const float* bv = (const float*)d_in[6];
  const float* Wo = (const float*)d_in[7];
  const float* bo = (const float*)d_in[8];
  float* out = (float*)d_out;

  // Workspace carve-up (bf16 elements, ~31 MiB total).
  bf16* p   = (bf16*)d_ws;
  bf16* Xs  = p; p += (long)S_LEN * DMODEL;           // packed A tiles of x
  bf16* Wqt = p; p += (long)NHEAD * DKH * DMODEL;     // [h][n][k]
  bf16* Wkt = p; p += (long)NHEAD * DKH * DMODEL;
  bf16* Wvt = p; p += (long)NHEAD * DVH * DMODEL;
  bf16* Wot = p; p += (long)DMODEL * (NHEAD * DVH);   // [n][k]
  bf16* Qb  = p; p += (long)NHEAD * S_LEN * DKH;      // row-major per head
  bf16* Kb  = p; p += (long)NHEAD * S_LEN * DKH;
  bf16* Vtw = p; p += (long)NHEAD * DVH * S_LEN;      // V transposed [h][dv][s]
  bf16* Ccp = p; p += (long)S_LEN * DMODEL;           // packed A tiles of concat

  // ---- casts / layout transforms -------------------------------------
  {
    long n = (long)S_LEN * DMODEL;
    cast_pack_a<<<(int)((n + 255) / 256), 256, 0, stream>>>(x, Xs, DMODEL, n);
  }
  {
    long n = (long)NHEAD * DMODEL * DKH;
    cast_transpose<<<(int)((n + 255) / 256), 256, 0, stream>>>(Wq, Wqt, DMODEL, DKH, n);
    cast_transpose<<<(int)((n + 255) / 256), 256, 0, stream>>>(Wk, Wkt, DMODEL, DKH, n);
  }
  {
    long n = (long)NHEAD * DMODEL * DVH;
    cast_transpose<<<(int)((n + 255) / 256), 256, 0, stream>>>(Wv, Wvt, DMODEL, DVH, n);
  }
  {
    long n = (long)DMODEL * (NHEAD * DVH);
    cast_transpose<<<(int)((n + 255) / 256), 256, 0, stream>>>(Wo, Wot, NHEAD * DVH,
                                                               DMODEL, n);
  }

  // ---- Q/K projections: [4096x512] @ [512x128] + b, per head ---------
  {
    dim3 grid((S_LEN / 64) * (DKH / 32) / 8, NHEAD);
    gemm_wmma<false, bf16><<<grid, 256, 0, stream>>>(
        Xs, Wqt, bq, Qb, S_LEN, DKH, DMODEL,
        (long)DKH * DMODEL, (long)S_LEN * DKH, DKH);
    gemm_wmma<false, bf16><<<grid, 256, 0, stream>>>(
        Xs, Wkt, bk, Kb, S_LEN, DKH, DMODEL,
        (long)DKH * DMODEL, (long)S_LEN * DKH, DKH);
  }
  // ---- V projection, stored transposed [h][dv][s] --------------------
  {
    dim3 grid((S_LEN / 64) * (DVH / 32) / 8, NHEAD);
    gemm_wmma<true, bf16><<<grid, 256, 0, stream>>>(
        Xs, Wvt, bv, Vtw, S_LEN, DVH, DMODEL,
        (long)DVH * DMODEL, (long)DVH * S_LEN, DVH);
  }
  // ---- Flash attention -> packed concat tiles ------------------------
  {
    dim3 grid(S_LEN / 128, NHEAD);
    flash_attn_wmma<<<grid, 256, 0, stream>>>(Qb, Kb, Vtw, Ccp);
  }
  // ---- Output projection: [4096x512] @ [512x512] + bo -> f32 ---------
  {
    dim3 grid((S_LEN / 64) * (DMODEL / 32) / 8, 1);
    gemm_wmma<false, float><<<grid, 256, 0, stream>>>(
        Ccp, Wot, bo, out, S_LEN, DMODEL, NHEAD * DVH, 0L, 0L, 0L);
  }
}
